// RNN_81372450390091
// MI455X (gfx1250) — matrix operand order repaired and checked by reading, not compiled, not run on hardware
//
#include <hip/hip_runtime.h>
#include <hip/hip_bf16.h>

// MI455X / gfx1250 wave32 WMMA implementation of the char+word bi-GRU tagger.
//
// Strategy:
//  - Weights are converted once per call from f32 to f16 and pre-swizzled into
//    WMMA B-fragment order (one contiguous 32B v16h per lane per 32x16 K-tile),
//    so the inner GEMM loop issues wide global_load_b128s out of L2.
//  - Each workgroup (256 thr = 8 wave32) owns a tile of 16 sequences and scans
//    all T steps with h resident in LDS. The A operand is the fused [x_t | h]
//    panel, multiplied against the original concatenated GRU weights with
//    v_wmma_f32_16x16x32_f16 (f32 accumulate).
//  - blockIdx.y selects forward/backward direction (backward = per-row
//    reversed time index, clip(len-1-t, 0, T-1), mask t < len).

typedef __attribute__((ext_vector_type(16))) _Float16 v16h;
typedef __attribute__((ext_vector_type(8)))  float    v8f;

// ---------------------------------------------------------------------------
// A-fragment gather from a row-major LDS panel As[16][ldk] (f16).
// 16-bit A 16x32 layout: lanes 0-15 rows M=0-15; VGPR v<4 -> K = 2v+8*half,
// v>=4 -> K = 16 + 2(v-4) + 8*half (pairs are contiguous -> ds_load_b32).
// ---------------------------------------------------------------------------
__device__ __forceinline__ v16h load_a_frag(const _Float16* As, int ldk,
                                            int kt, int lane) {
  const int m = lane & 15;
  const int half = lane >> 4;
  const _Float16* base = As + m * ldk + kt * 32;
  v16h a;
#pragma unroll
  for (int v = 0; v < 8; ++v) {
    int k = (v < 4) ? (2 * v + 8 * half) : (16 + 2 * (v - 4) + 8 * half);
    a[2 * v]     = base[k];
    a[2 * v + 1] = base[k + 1];
  }
  return a;
}

// ---------------------------------------------------------------------------
// Weight pack: f32 row-major W[K][N] -> f16 B-fragments.
// B 32x16 layout (matches A): VGPR v<4 -> K = 2v+8*half, v>=4 -> K = 16 +
// 2(v-4) + 8*half (pairs K,K+1). Each lane's 16 halves stored contiguously (32B) so the scan
// kernel loads one v16h per fragment.
// ---------------------------------------------------------------------------
__global__ void pack_b_kernel(const float* __restrict__ W,
                              _Float16* __restrict__ out, int K, int N) {
  int idx = blockIdx.x * blockDim.x + threadIdx.x;
  int ntiles = N >> 4;
  int total = (K >> 5) * ntiles * 32;
  if (idx >= total) return;
  int lane = idx & 31;
  int tile = idx >> 5;
  int nt = tile % ntiles;
  int kt = tile / ntiles;
  int n = nt * 16 + (lane & 15);
  int half = lane >> 4;
  _Float16* dst = out + (size_t)idx * 16;
#pragma unroll
  for (int v = 0; v < 8; ++v) {
    int k0 = kt * 32 + ((v < 4) ? (2 * v + 8 * half) : (16 + 2 * (v - 4) + 8 * half));
    dst[2 * v]     = (_Float16)W[(size_t)k0 * N + n];
    dst[2 * v + 1] = (_Float16)W[(size_t)(k0 + 1) * N + n];
  }
}

// ---------------------------------------------------------------------------
// Fused GRU scan. XD = input width (128 = char mode, 1280 = word mode).
// Weights WgF: [KD][1024] (fragment-packed), WcF: [KD][512], KD = XD + 512.
// ---------------------------------------------------------------------------
template <int XD>
__global__ void __launch_bounds__(256) gru_scan_kernel(
    int T, const int* __restrict__ lens,
    // char mode inputs
    const int* __restrict__ chr_seqs, const float* __restrict__ chr_emb,
    // word mode inputs
    const int* __restrict__ cw_ids, const int* __restrict__ w_ids,
    const _Float16* __restrict__ cw_states, const float* __restrict__ w_emb,
    // shared
    const v16h* __restrict__ WgF, const v16h* __restrict__ WcF,
    const float* __restrict__ bg, const float* __restrict__ bc,
    _Float16* __restrict__ out16, float* __restrict__ out32) {
  constexpr int HD = 512;
  constexpr int GD = 1024;
  constexpr int KD = XD + HD;
  constexpr int KT = KD / 32;
  constexpr int NTG = GD / 16;  // 64 gate n-tiles
  constexpr int NTC = HD / 16;  // 32 candidate n-tiles

  extern __shared__ _Float16 smem[];
  _Float16* As = smem;             // [16][KD]  fused [x_t | h] (later [x_t | r*h])
  _Float16* hs = smem + 16 * KD;   // [16][HD]  hidden state
  _Float16* us = hs + 16 * HD;     // [16][HD]  update gate
  __shared__ int len_s[16];

  const int row0 = blockIdx.x * 16;
  const int dir = blockIdx.y;      // 0 = forward, 1 = backward
  const int tid = threadIdx.x;
  const int wave = tid >> 5;
  const int lane = tid & 31;

  for (int i = tid; i < 16 * HD; i += 256) hs[i] = (_Float16)0.0f;
  if (tid < 16) len_s[tid] = lens[row0 + tid];
  __syncthreads();

  const v8f zero8 = {};

  for (int t = 0; t < T; ++t) {
    // ---- stage x_t into As[:, 0:XD] (gather + f32->f16) -------------------
    for (int i = tid; i < 16 * XD; i += 256) {
      int r = i / XD;
      int c = i - r * XD;
      int tsrc = t;
      if (dir) {
        tsrc = len_s[r] - 1 - t;
        tsrc = tsrc < 0 ? 0 : tsrc;
        tsrc = tsrc > T - 1 ? T - 1 : tsrc;
      }
      float x;
      if constexpr (XD == 128) {
        int ch = chr_seqs[(row0 + r) * T + tsrc];
        x = chr_emb[ch * 128 + c];
      } else {
        if (c < 1024) {
          int u = cw_ids[(row0 + r) * T + tsrc];
          x = (float)cw_states[(size_t)u * 1024 + c];
        } else {
          int w = w_ids[(row0 + r) * T + tsrc];
          x = w_emb[(size_t)w * 256 + (c - 1024)];
        }
      }
      As[r * KD + c] = (_Float16)x;
    }
    // ---- stage h into As[:, XD:KD] ---------------------------------------
    for (int i = tid; i < 16 * HD; i += 256) {
      int r = i / HD;
      int c = i - r * HD;
      As[r * KD + XD + c] = hs[i];
    }
    __syncthreads();

    // ---- phase 1: gates = sigmoid([x,h] @ Wg + bg), 8 n-tiles per wave ----
    v8f g[8];
#pragma unroll
    for (int j = 0; j < 8; ++j) g[j] = zero8;
    for (int kt = 0; kt < KT; ++kt) {
      v16h a = load_a_frag(As, KD, kt, lane);
#pragma unroll
      for (int j = 0; j < 8; ++j) {
        int nt = wave * 8 + j;
        v16h b = WgF[((size_t)kt * NTG + nt) * 32 + lane];
        g[j] = __builtin_amdgcn_wmma_f32_16x16x32_f16(
            false, a, false, b, (short)0, g[j], false, false);
      }
    }
    __syncthreads();  // all waves done reading As before r*h overwrites it

#pragma unroll
    for (int j = 0; j < 8; ++j) {
      int nt = wave * 8 + j;
      int n = nt * 16 + (lane & 15);
      int half = lane >> 4;
#pragma unroll
      for (int v = 0; v < 8; ++v) {
        int m = v + 8 * half;
        float gv = g[j][v] + bg[n];
        gv = 1.0f / (1.0f + expf(-gv));
        if (n < HD) {                      // reset gate -> stage r*h
          float h = (float)hs[m * HD + n];
          As[m * KD + XD + n] = (_Float16)(gv * h);
        } else {                           // update gate
          us[m * HD + (n - HD)] = (_Float16)gv;
        }
      }
    }
    __syncthreads();

    // ---- phase 2: c = tanh([x, r*h] @ Wc + bc); h = u*h + (1-u)*c ---------
    v8f cc[4];
#pragma unroll
    for (int j = 0; j < 4; ++j) cc[j] = zero8;
    for (int kt = 0; kt < KT; ++kt) {
      v16h a = load_a_frag(As, KD, kt, lane);
#pragma unroll
      for (int j = 0; j < 4; ++j) {
        int nt = wave * 4 + j;
        v16h b = WcF[((size_t)kt * NTC + nt) * 32 + lane];
        cc[j] = __builtin_amdgcn_wmma_f32_16x16x32_f16(
            false, a, false, b, (short)0, cc[j], false, false);
      }
    }
#pragma unroll
    for (int j = 0; j < 4; ++j) {
      int nt = wave * 4 + j;
      int n = nt * 16 + (lane & 15);
      int half = lane >> 4;
#pragma unroll
      for (int v = 0; v < 8; ++v) {
        int m = v + 8 * half;
        float c = tanhf(cc[j][v] + bc[n]);
        float u = (float)us[m * HD + n];
        float h = (float)hs[m * HD + n];
        float hn = u * h + (1.0f - u) * c;
        hs[m * HD + n] = (_Float16)((t < len_s[m]) ? hn : h);
      }
    }
    __syncthreads();
  }

  // ---- write final state: fw -> cols [0,512), bw -> cols [512,1024) -------
  for (int i = tid; i < 16 * HD; i += 256) {
    int r = i / HD;
    int c = i - r * HD;
    if constexpr (XD == 128) {
      out16[(size_t)(row0 + r) * 1024 + dir * HD + c] = hs[i];
    } else {
      out32[(size_t)(row0 + r) * 1024 + dir * HD + c] = (float)hs[i];
    }
  }
}

// ---------------------------------------------------------------------------
// FC head: leaky_relu(states @ W1 + b1, 0.2) @ W2 + b2. Tiny; scalar f32.
// ---------------------------------------------------------------------------
__global__ void head_kernel(const float* __restrict__ states,
                            const float* __restrict__ W1,
                            const float* __restrict__ b1,
                            const float* __restrict__ W2,
                            const float* __restrict__ b2,
                            float* __restrict__ out) {
  __shared__ float hid[64];
  int b = blockIdx.x;
  int j = threadIdx.x;
  float s = 0.0f;
  for (int k = 0; k < 1024; ++k) s += states[(size_t)b * 1024 + k] * W1[k * 64 + j];
  s += b1[j];
  hid[j] = (s > 0.0f) ? s : 0.2f * s;
  __syncthreads();
  if (j < 2) {
    float o = b2[j];
    for (int k = 0; k < 64; ++k) o += hid[k] * W2[k * 2 + j];
    out[b * 2 + j] = o;
  }
}

// ---------------------------------------------------------------------------
extern "C" void kernel_launch(void* const* d_in, const int* in_sizes, int n_in,
                              void* d_out, int out_size, void* d_ws, size_t ws_size,
                              hipStream_t stream) {
  (void)in_sizes; (void)n_in; (void)out_size; (void)ws_size;

  const int*   charseqs      = (const int*)d_in[0];   // [2048][16]
  const int*   charseq_lens  = (const int*)d_in[1];   // [2048]
  const int*   charseq_ids   = (const int*)d_in[2];   // [256][64]
  const int*   word_ids      = (const int*)d_in[3];   // [256][64]
  const int*   sentence_lens = (const int*)d_in[4];   // [256]
  const float* char_emb      = (const float*)d_in[5]; // [256][128]
  const float* word_emb      = (const float*)d_in[6]; // [100000][256]
  const float* Wg_c          = (const float*)d_in[7]; // [640][1024]
  const float* bg_c          = (const float*)d_in[8];
  const float* Wc_c          = (const float*)d_in[9]; // [640][512]
  const float* bc_c          = (const float*)d_in[10];
  const float* Wg_w          = (const float*)d_in[11]; // [1792][1024]
  const float* bg_w          = (const float*)d_in[12];
  const float* Wc_w          = (const float*)d_in[13]; // [1792][512]
  const float* bc_w          = (const float*)d_in[14];
  const float* W1            = (const float*)d_in[15]; // [1024][64]
  const float* b1            = (const float*)d_in[16];
  const float* W2            = (const float*)d_in[17]; // [64][2]
  const float* b2            = (const float*)d_in[18];
  float* out = (float*)d_out;                          // [256][2]

  // Workspace carve-up (256B aligned).
  char* ws = (char*)d_ws;
  size_t off = 0;
  auto carve = [&](size_t bytes) -> void* {
    void* p = ws + off;
    off = (off + bytes + 255) & ~(size_t)255;
    return p;
  };
  _Float16* WgcF = (_Float16*)carve((size_t)640 * 1024 * 2);
  _Float16* WccF = (_Float16*)carve((size_t)640 * 512 * 2);
  _Float16* WgwF = (_Float16*)carve((size_t)1792 * 1024 * 2);
  _Float16* WcwF = (_Float16*)carve((size_t)1792 * 512 * 2);
  _Float16* char_states = (_Float16*)carve((size_t)2048 * 1024 * 2); // f16 [U][1024]
  float*    word_states = (float*)carve((size_t)256 * 1024 * 4);     // f32 [B][1024]

  // 1) Pack weights into WMMA B-fragment layout (f32 -> f16).
  auto packN = [](int K, int N) { return (K >> 5) * (N >> 4) * 32; };
  pack_b_kernel<<<(packN(640, 1024) + 255) / 256, 256, 0, stream>>>(Wg_c, WgcF, 640, 1024);
  pack_b_kernel<<<(packN(640, 512)  + 255) / 256, 256, 0, stream>>>(Wc_c, WccF, 640, 512);
  pack_b_kernel<<<(packN(1792, 1024) + 255) / 256, 256, 0, stream>>>(Wg_w, WgwF, 1792, 1024);
  pack_b_kernel<<<(packN(1792, 512)  + 255) / 256, 256, 0, stream>>>(Wc_w, WcwF, 1792, 512);

  // 2) Char bi-GRU: 2048 rows / 16 = 128 tiles, y = direction.
  {
    dim3 grid(2048 / 16, 2);
    size_t shmem = (size_t)(16 * (128 + 512) + 2 * 16 * 512) * sizeof(_Float16); // 52 KB
    gru_scan_kernel<128><<<grid, 256, shmem, stream>>>(
        16, charseq_lens,
        charseqs, char_emb,
        nullptr, nullptr, nullptr, nullptr,
        (const v16h*)WgcF, (const v16h*)WccF, bg_c, bc_c,
        char_states, nullptr);
  }

  // 3) Word bi-GRU: 256 rows / 16 = 16 tiles, y = direction.
  {
    dim3 grid(256 / 16, 2);
    size_t shmem = (size_t)(16 * (1280 + 512) + 2 * 16 * 512) * sizeof(_Float16); // 88 KB
    gru_scan_kernel<1280><<<grid, 256, shmem, stream>>>(
        64, sentence_lens,
        nullptr, nullptr,
        charseq_ids, word_ids, char_states, word_emb,
        (const v16h*)WgwF, (const v16h*)WcwF, bg_w, bc_w,
        nullptr, word_states);
  }

  // 4) FC head.
  head_kernel<<<256, 64, 0, stream>>>(word_states, W1, b1, W2, b2, out);
}
